// DecoderCrossAttn_82145544503487
// MI455X (gfx1250) — matrix operand order repaired and checked by reading, not compiled, not run on hardware
//
#include <hip/hip_runtime.h>

// ---------------------------------------------------------------------------
// DecoderCrossAttn for MI455X (gfx1250): Mamba + cross-attention stack.
// All GEMMs run on v_wmma_f32_16x16x32_f16 (f16 in, f32 accum).
// GEMM: branch-free (EXEC all-ones around WMMA), block tile 64x128, 4 waves,
// wave tile 32x64 = 8 WMMAs per K-step over 6 fragment loads.
// All GEMM dims padded to M%64==0, N%128==0, K%32==0.
// Workspace requirement: ~118 MB in d_ws.
// ---------------------------------------------------------------------------

constexpr int B_    = 2;
constexpr int L_    = 1024;
constexpr int LE_   = 1024;
constexpr int NM_   = 80;
constexpr int DM_   = 512;
constexpr int DI_   = 1024;   // EXP * DM
constexpr int DTR_  = 32;     // (DM+15)/16
constexpr int DS_   = 16;
constexpr int H_    = 4;
constexpr int HD_   = 128;    // DM / H
constexpr int LAYERS_ = 4;
constexpr int KC_   = 4;      // conv kernel width
constexpr int ME_   = B_ * L_;    // 2048 decoder tokens
constexpr int MENC_ = B_ * LE_;   // 2048 encoder tokens
constexpr int NMP_  = 96;         // NM padded to K%32==0 (as GEMM K)
constexpr int NOP_  = 128;        // NM padded to N%128==0 (as GEMM N)
constexpr int XDL_  = 128;        // x_dbl leading dim (64 padded to N%128==0)
constexpr int LDXR_ = 2 * DI_;    // 2048

typedef __attribute__((ext_vector_type(16))) _Float16 v16h;
typedef __attribute__((ext_vector_type(8)))  float    v8f;
typedef __attribute__((ext_vector_type(4)))  float    v4f;

union Frag16 { v16h v; v4f f[2]; };

// A fragment: 16x32 (MxK) f16 tile, row-major memory, ld in elements.
// ISA layout: lanes 0-15 -> row=lane, halves[0..7]=K 0..7, halves[8..15]=K 16..23
//             lanes 16-31 -> row=lane-16, halves[0..7]=K 8..15, halves[8..15]=K 24..31
__device__ __forceinline__ v16h load_a_16x32(const _Float16* __restrict__ base, int ld) {
  const int lane = threadIdx.x & 31;
  const int r    = lane & 15;
  const int kh   = (lane >> 4) << 3;          // 0 or 8
  const _Float16* p = base + (long)r * ld + kh;
  Frag16 fr;
  fr.f[0] = *(const v4f*)(p);                 // K kh..kh+7
  fr.f[1] = *(const v4f*)(p + 16);            // K kh+16..kh+23
  return fr.v;
}

// B fragment: 32x16 (KxN) where memory holds W[N,K] row-major (so B = W^T tile).
// ISA layout (16b B-matrix tables): lanes 0-15 -> col=lane, K 0..7, 16..23;
//                                   lanes 16-31 -> col=lane-16, K 8..15, 24..31.
__device__ __forceinline__ v16h load_b_32x16(const _Float16* __restrict__ base, int ld) {
  const int lane = threadIdx.x & 31;
  const int c    = lane & 15;
  const int kh   = (lane >> 4) << 3;          // 0 or 8
  const _Float16* p = base + (long)c * ld + kh;
  Frag16 fr;
  fr.f[0] = *(const v4f*)(p);                 // K kh..kh+7
  fr.f[1] = *(const v4f*)(p + 16);            // K kh+16..kh+23
  return fr.v;
}

// C/D 16x16 f32 tile store. VGPR e: lanes 0-15 -> (M=e, N=lane); lanes 16-31 -> (M=e+8).
__device__ __forceinline__ void store_tile(float* __restrict__ C, _Float16* __restrict__ C16,
                                           const float* __restrict__ bias, int ldc,
                                           long row0, long col0, float alpha, v8f d) {
  const int lane = threadIdx.x & 31;
  const int cl = lane & 15, chh = lane >> 4;
  const float bv = bias ? bias[col0 + cl] : 0.f;
#pragma unroll
  for (int e = 0; e < 8; ++e) {
    const long off = (row0 + e + 8 * chh) * (long)ldc + col0 + cl;
    const float val = d[e] * alpha + bv;
    C[off] = val;
    if (C16) C16[off] = (_Float16)val;
  }
}

// ---------------------------------------------------------------------------
// Generic batched GEMM: C[z] = alpha * A[z] x B[z]^T + bias
//   A: [M,K] f16 (lda), B: [N,K] f16 (ldb), C: [M,N] f32 (ldc), optional f16 copy.
// Batch z decomposed as zo = z / innerCnt, zi = z % innerCnt with separate strides
// (element units) so attention heads (non-linear offsets) are expressible.
// Block: 128 threads = 4 waves in 2x2; wave tile 32x64 (2 A-frags x 4 B-frags
// -> 8 WMMAs per K-step). Block tile 64x128.
// REQUIRES: K % 32 == 0, M % 64 == 0, N % 128 == 0 (no divergent guards ->
// EXEC stays all-ones around every WMMA, per ISA requirement).
// ---------------------------------------------------------------------------
__global__ __launch_bounds__(128)
void gemm_abt_wmma(const _Float16* __restrict__ A, const _Float16* __restrict__ Bw,
                   const float* __restrict__ bias, float* __restrict__ C,
                   _Float16* __restrict__ C16,
                   int M, int N, int K, int lda, int ldb, int ldc, float alpha,
                   long aOut, long aIn, long bOut, long bIn, long cOut, long cIn,
                   int innerCnt) {
  const int z  = blockIdx.z;
  const int zo = z / innerCnt;
  const int zi = z % innerCnt;
  A  += (long)zo * aOut + (long)zi * aIn;
  Bw += (long)zo * bOut + (long)zi * bIn;
  const long coff = (long)zo * cOut + (long)zi * cIn;
  C += coff;
  if (C16) C16 += coff;            // scalar (kernel-arg) condition: no divergence

  const int wave = threadIdx.x >> 5;           // 0..3
  const int wm = wave >> 1, wn = wave & 1;
  const int m0 = blockIdx.y * 64 + wm * 32;    // 2 M-subtiles: m0, m0+16
  const int n0 = blockIdx.x * 128 + wn * 64;   // 4 N-subtiles: n0 + 16*j

  const _Float16* Arow0 = A + (long)m0 * lda;
  const _Float16* Arow1 = A + (long)(m0 + 16) * lda;
  const _Float16* Brow[4];
#pragma unroll
  for (int j = 0; j < 4; ++j) Brow[j] = Bw + (long)(n0 + 16 * j) * ldb;

  v8f acc0[4], acc1[4];
#pragma unroll
  for (int j = 0; j < 4; ++j) {
    acc0[j] = (v8f){0.f,0.f,0.f,0.f,0.f,0.f,0.f,0.f};
    acc1[j] = (v8f){0.f,0.f,0.f,0.f,0.f,0.f,0.f,0.f};
  }

  const int lane = threadIdx.x & 31;
  for (int k0 = 0; k0 < K; k0 += 32) {
    v16h a0 = load_a_16x32(Arow0 + k0, lda);
    v16h a1 = load_a_16x32(Arow1 + k0, lda);
    v16h b[4];
#pragma unroll
    for (int j = 0; j < 4; ++j) b[j] = load_b_32x16(Brow[j] + k0, ldb);

    // Pull the next K-tile toward the WGP while the WMMAs below execute.
    // (Speculative prefetch: OOB translations are silently dropped.)
    __builtin_prefetch(Arow0 + (long)(lane & 15) * lda + k0 + 32, 0, 1);
    __builtin_prefetch(Brow[0] + (long)(lane & 15) * ldb + k0 + 32, 0, 1);

#pragma unroll
    for (int j = 0; j < 4; ++j) {
      acc0[j] = __builtin_amdgcn_wmma_f32_16x16x32_f16(false, a0, false, b[j], (short)0, acc0[j], false, false);
      acc1[j] = __builtin_amdgcn_wmma_f32_16x16x32_f16(false, a1, false, b[j], (short)0, acc1[j], false, false);
    }
  }

#pragma unroll
  for (int j = 0; j < 4; ++j) {
    store_tile(C, C16, bias, ldc, m0,      n0 + 16 * j, alpha, acc0[j]);
    store_tile(C, C16, bias, ldc, m0 + 16, n0 + 16 * j, alpha, acc1[j]);
  }
}

// ---------------------------------------------------------------------------
// Elementwise / reduction kernels (f32 math, f16 emission for GEMM operands)
// ---------------------------------------------------------------------------

// f32 -> f16 with row stride change + zero padding in both cols (K-pad) and
// rows (N-pad). total = drows * dld; source valid region is srows x cols.
__global__ void convert_pad_kernel(const float* __restrict__ src, int sld, int cols, int srows,
                                   _Float16* __restrict__ dst, int dld, long total) {
  long i = blockIdx.x * (long)blockDim.x + threadIdx.x;
  if (i >= total) return;
  int  c = (int)(i % dld);
  long r = i / dld;
  dst[i] = (c < cols && r < srows) ? (_Float16)src[r * (long)sld + c] : (_Float16)0.f;
}

// copy valid 80 columns out of the 128-padded final GEMM result
__global__ void copy_out_kernel(const float* __restrict__ src, float* __restrict__ dst, long total) {
  long i = blockIdx.x * (long)blockDim.x + threadIdx.x;
  if (i >= total) return;
  const long r = i / NM_;
  const int  c = (int)(i % NM_);
  dst[i] = src[r * NOP_ + c];
}

// rmsnorm over rows of DM_, f16 output
__global__ __launch_bounds__(256)
void rmsnorm_kernel(const float* __restrict__ x, const float* __restrict__ w,
                    _Float16* __restrict__ out) {
  __shared__ float red[256];
  const long row = blockIdx.x;
  const float* xr = x + row * DM_;
  float local[DM_ / 256];
  float s = 0.f;
  for (int c = threadIdx.x, j = 0; c < DM_; c += 256, ++j) { local[j] = xr[c]; s += local[j] * local[j]; }
  red[threadIdx.x] = s; __syncthreads();
  for (int st = 128; st > 0; st >>= 1) { if (threadIdx.x < st) red[threadIdx.x] += red[threadIdx.x + st]; __syncthreads(); }
  const float inv = rsqrtf(red[0] / DM_ + 1e-5f);
  for (int c = threadIdx.x, j = 0; c < DM_; c += 256, ++j)
    out[row * DM_ + c] = (_Float16)(local[j] * inv * w[c]);
}

// causal depthwise conv (K=4) + bias + SiLU; input = xm part of xr [ME, 2*DI]
__global__ void conv_silu_kernel(const float* __restrict__ xr, const float* __restrict__ w,
                                 const float* __restrict__ bias, float* __restrict__ u,
                                 _Float16* __restrict__ u16) {
  long idx = blockIdx.x * (long)blockDim.x + threadIdx.x;
  if (idx >= (long)ME_ * DI_) return;
  const int  c  = (int)(idx % DI_);
  const long bl = idx / DI_;
  const int  l  = (int)(bl % L_);
  float acc = bias[c];
#pragma unroll
  for (int j = 0; j < KC_; ++j) {
    const int lj = l - (KC_ - 1) + j;
    if (lj >= 0) acc += xr[(bl - (KC_ - 1) + j) * (long)LDXR_ + c] * w[c * KC_ + j];
  }
  const float s = acc / (1.f + __expf(-acc));
  u[idx] = s;
  u16[idx] = (_Float16)s;
}

__global__ void softplus_kernel(float* __restrict__ d, long n) {
  long i = blockIdx.x * (long)blockDim.x + threadIdx.x;
  if (i >= n) return;
  const float x = d[i];
  d[i] = (x > 20.f) ? x : log1pf(__expf(x));
}

// selective scan: recurrence x_t = exp(delta*A)*x_{t-1} + delta*u*B ; y = <x,C> + u*D
// one 16-lane group per (b, channel); lanes = states; epilogue applies SiLU(res) gate.
__global__ __launch_bounds__(128)
void scan_kernel(const float* __restrict__ delta, const float* __restrict__ u,
                 const float* __restrict__ xdbl, const float* __restrict__ xr,
                 const float* __restrict__ Alog, const float* __restrict__ Dv,
                 _Float16* __restrict__ y16) {
  const int grp = blockIdx.x * (blockDim.x >> 4) + (threadIdx.x >> 4);
  const int n   = threadIdx.x & 15;
  if (grp >= B_ * DI_) return;
  const int b = grp / DI_, d = grp % DI_;
  const float A   = -__expf(Alog[d * DS_ + n]);
  const float Dvv = Dv[d];
  float x = 0.f;
  for (int t = 0; t < L_; ++t) {
    const long base = (long)b * L_ + t;
    const float dl = delta[base * DI_ + d];
    const float uu = u[base * DI_ + d];
    const float Bn = xdbl[base * XDL_ + DTR_ + n];
    const float Cn = xdbl[base * XDL_ + DTR_ + DS_ + n];
    x = __expf(dl * A) * x + dl * uu * Bn;
    float p = x * Cn;
    p += __shfl_xor(p, 1, 16);
    p += __shfl_xor(p, 2, 16);
    p += __shfl_xor(p, 4, 16);
    p += __shfl_xor(p, 8, 16);
    if (n == 0) {
      const float res  = xr[base * (long)LDXR_ + DI_ + d];
      const float gate = res / (1.f + __expf(-res));
      y16[base * DI_ + d] = (_Float16)((p + uu * Dvv) * gate);
    }
  }
}

__global__ void add_residual_kernel(float* __restrict__ h, const float* __restrict__ a,
                                    _Float16* __restrict__ h16, long n) {
  long i = blockIdx.x * (long)blockDim.x + threadIdx.x;
  if (i >= n) return;
  const float v = h[i] + a[i];
  h[i] = v;
  h16[i] = (_Float16)v;
}

// h = layernorm(h + o) * w + b ; writes f32 + f16
__global__ __launch_bounds__(256)
void add_layernorm_kernel(float* __restrict__ h, const float* __restrict__ o,
                          const float* __restrict__ w, const float* __restrict__ b,
                          _Float16* __restrict__ h16) {
  __shared__ float red[256];
  const long row = blockIdx.x;
  float* hr = h + row * DM_;
  const float* orr = o + row * DM_;
  float local[DM_ / 256];
  float s = 0.f;
  for (int c = threadIdx.x, j = 0; c < DM_; c += 256, ++j) { local[j] = hr[c] + orr[c]; s += local[j]; }
  red[threadIdx.x] = s; __syncthreads();
  for (int st = 128; st > 0; st >>= 1) { if (threadIdx.x < st) red[threadIdx.x] += red[threadIdx.x + st]; __syncthreads(); }
  const float mean = red[0] / DM_;
  __syncthreads();
  float vs = 0.f;
  for (int j = 0; j < DM_ / 256; ++j) { const float dd = local[j] - mean; vs += dd * dd; }
  red[threadIdx.x] = vs; __syncthreads();
  for (int st = 128; st > 0; st >>= 1) { if (threadIdx.x < st) red[threadIdx.x] += red[threadIdx.x + st]; __syncthreads(); }
  const float inv = rsqrtf(red[0] / DM_ + 1e-5f);
  for (int c = threadIdx.x, j = 0; c < DM_; c += 256, ++j) {
    const float val = (local[j] - mean) * inv * w[c] + b[c];
    hr[c] = val;
    h16[row * DM_ + c] = (_Float16)val;
  }
}

// softmax row-wise, f32 -> f16
__global__ __launch_bounds__(256)
void softmax_kernel(const float* __restrict__ S, _Float16* __restrict__ P, int cols) {
  __shared__ float red[256];
  const long row = blockIdx.x;
  const float* s = S + row * (long)cols;
  _Float16* p = P + row * (long)cols;
  float mx = -1e30f;
  for (int c = threadIdx.x; c < cols; c += 256) mx = fmaxf(mx, s[c]);
  red[threadIdx.x] = mx; __syncthreads();
  for (int st = 128; st > 0; st >>= 1) { if (threadIdx.x < st) red[threadIdx.x] = fmaxf(red[threadIdx.x], red[threadIdx.x + st]); __syncthreads(); }
  mx = red[0]; __syncthreads();
  float sum = 0.f;
  for (int c = threadIdx.x; c < cols; c += 256) sum += __expf(s[c] - mx);
  red[threadIdx.x] = sum; __syncthreads();
  for (int st = 128; st > 0; st >>= 1) { if (threadIdx.x < st) red[threadIdx.x] += red[threadIdx.x + st]; __syncthreads(); }
  const float inv = 1.f / red[0];
  for (int c = threadIdx.x; c < cols; c += 256) p[c] = (_Float16)(__expf(s[c] - mx) * inv);
}

// vt[b,h,d,k] = v[b,k, h*HD + d]  (f32 -> f16), so P @ V reuses the A x B^T GEMM
__global__ void transpose_v_kernel(const float* __restrict__ v, _Float16* __restrict__ vt) {
  long idx = blockIdx.x * (long)blockDim.x + threadIdx.x;
  if (idx >= (long)B_ * H_ * HD_ * LE_) return;
  const int  k  = (int)(idx % LE_);
  const long r  = idx / LE_;
  const int  dd = (int)(r % HD_);
  const long bh = r / HD_;
  const int  hh = (int)(bh % H_);
  const int  bb = (int)(bh / H_);
  vt[idx] = (_Float16)v[((long)bb * LE_ + k) * DM_ + hh * HD_ + dd];
}

// ---------------------------------------------------------------------------
// Host orchestration
// ---------------------------------------------------------------------------
static void launch_gemm(hipStream_t s, const _Float16* A, const _Float16* Bw,
                        const float* bias, float* C, _Float16* C16,
                        int M, int N, int K, int lda, int ldb, int ldc, float alpha,
                        int nbat = 1, int innerCnt = 1,
                        long aOut = 0, long aIn = 0, long bOut = 0, long bIn = 0,
                        long cOut = 0, long cIn = 0) {
  // Requires M % 64 == 0, N % 128 == 0, K % 32 == 0.
  dim3 g((unsigned)(N / 128), (unsigned)(M / 64), (unsigned)nbat);
  gemm_abt_wmma<<<g, 128, 0, s>>>(A, Bw, bias, C, C16, M, N, K, lda, ldb, ldc, alpha,
                                  aOut, aIn, bOut, bIn, cOut, cIn, innerCnt < 1 ? 1 : innerCnt);
}

extern "C" void kernel_launch(void* const* d_in, const int* in_sizes, int n_in,
                              void* d_out, int out_size, void* d_ws, size_t ws_size,
                              hipStream_t stream) {
  (void)in_sizes; (void)n_in; (void)out_size; (void)ws_size;
  const float* x        = (const float*)d_in[0];
  const float* enc      = (const float*)d_in[1];
  const float* W_in     = (const float*)d_in[2];
  const float* b_in     = (const float*)d_in[3];
  const float* m_norm_w = (const float*)d_in[4];
  const float* m_in_w   = (const float*)d_in[5];
  const float* m_conv_w = (const float*)d_in[6];
  const float* m_conv_b = (const float*)d_in[7];
  const float* m_xproj  = (const float*)d_in[8];
  const float* m_dt_w   = (const float*)d_in[9];
  const float* m_dt_b   = (const float*)d_in[10];
  const float* m_Alog   = (const float*)d_in[11];
  const float* m_D      = (const float*)d_in[12];
  const float* m_out_w  = (const float*)d_in[13];
  const float* a_in_w   = (const float*)d_in[14];
  const float* a_in_b   = (const float*)d_in[15];
  const float* a_out_w  = (const float*)d_in[16];
  const float* a_out_b  = (const float*)d_in[17];
  const float* a_ln_w   = (const float*)d_in[18];
  const float* a_ln_b   = (const float*)d_in[19];
  const float* normf_w  = (const float*)d_in[20];
  const float* W_out    = (const float*)d_in[21];
  float* out = (float*)d_out;

  // ---- workspace carve (bump allocator, 256B aligned) ----
  char* p = (char*)d_ws;
  auto take = [&](size_t bytes) { void* r = (void*)p; p += (bytes + 255) & ~(size_t)255; return r; };
  float*    hf32  = (float*)   take((size_t)ME_ * DM_ * 4);
  _Float16* hf16  = (_Float16*)take((size_t)ME_ * DM_ * 2);
  _Float16* hn16  = (_Float16*)take((size_t)ME_ * DM_ * 2);
  _Float16* enc16 = (_Float16*)take((size_t)MENC_ * DM_ * 2);
  _Float16* xf16  = (_Float16*)take((size_t)ME_ * NMP_ * 2);
  _Float16* wbuf  = (_Float16*)take((size_t)2048 * 512 * 2);
  float*    xr    = (float*)   take((size_t)ME_ * LDXR_ * 4);
  float*    u     = (float*)   take((size_t)ME_ * DI_ * 4);
  _Float16* u16   = (_Float16*)take((size_t)ME_ * DI_ * 2);
  float*    xdbl  = (float*)   take((size_t)ME_ * XDL_ * 4);
  _Float16* dlt16 = (_Float16*)take((size_t)ME_ * DTR_ * 2);
  float*    delta = (float*)   take((size_t)ME_ * DI_ * 4);
  _Float16* y16   = (_Float16*)take((size_t)ME_ * DI_ * 2);
  float*    mo    = (float*)   take((size_t)ME_ * DM_ * 4);
  float*    tmp   = (float*)   take((size_t)ME_ * DM_ * 4);
  _Float16* q16   = (_Float16*)take((size_t)ME_ * DM_ * 2);
  _Float16* k16   = (_Float16*)take((size_t)MENC_ * DM_ * 2);
  _Float16* vt16  = (_Float16*)take((size_t)B_ * H_ * HD_ * LE_ * 2);
  float*    S     = (float*)   take((size_t)B_ * H_ * L_ * LE_ * 4);
  _Float16* P16   = (_Float16*)take((size_t)B_ * H_ * L_ * LE_ * 2);
  _Float16* o16   = (_Float16*)take((size_t)ME_ * DM_ * 2);

  auto cvt = [&](const float* s, int sld, int cols, int srows, _Float16* d, int dld, long drows) {
    long tot = drows * (long)dld;
    convert_pad_kernel<<<(unsigned)((tot + 255) / 256), 256, 0, stream>>>(s, sld, cols, srows, d, dld, tot);
  };

  // ---- input projection: h = x @ W_in^T + b_in  (K padded 80 -> 96) ----
  cvt(enc, DM_, DM_, MENC_, enc16, DM_, MENC_);
  cvt(x, NM_, NM_, ME_, xf16, NMP_, ME_);
  cvt(W_in, NM_, NM_, DM_, wbuf, NMP_, DM_);
  launch_gemm(stream, xf16, wbuf, b_in, hf32, hf16, ME_, DM_, NMP_, NMP_, NMP_, DM_, 1.f);

  const float attn_scale = 1.f / sqrtf((float)HD_);

  for (int i = 0; i < LAYERS_; ++i) {
    // ---- Mamba block ----
    rmsnorm_kernel<<<ME_, 256, 0, stream>>>(hf32, m_norm_w + (long)i * DM_, hn16);
    cvt(m_in_w + (long)i * 2 * DI_ * DM_, DM_, DM_, 2 * DI_, wbuf, DM_, 2 * DI_);
    launch_gemm(stream, hn16, wbuf, nullptr, xr, nullptr, ME_, 2 * DI_, DM_, DM_, DM_, LDXR_, 1.f);

    conv_silu_kernel<<<(ME_ * DI_ + 255) / 256, 256, 0, stream>>>(
        xr, m_conv_w + (long)i * DI_ * KC_, m_conv_b + (long)i * DI_, u, u16);

    // xproj: N padded 64 -> 128 (weight rows 64..127 zero, x_dbl ld = 128)
    cvt(m_xproj + (long)i * 64 * DI_, DI_, DI_, 64, wbuf, DI_, XDL_);
    launch_gemm(stream, u16, wbuf, nullptr, xdbl, nullptr, ME_, XDL_, DI_, DI_, DI_, XDL_, 1.f);

    cvt(xdbl, XDL_, DTR_, ME_, dlt16, DTR_, ME_);
    cvt(m_dt_w + (long)i * DI_ * DTR_, DTR_, DTR_, DI_, wbuf, DTR_, DI_);
    launch_gemm(stream, dlt16, wbuf, m_dt_b + (long)i * DI_, delta, nullptr,
                ME_, DI_, DTR_, DTR_, DTR_, DI_, 1.f);
    softplus_kernel<<<(ME_ * DI_ + 255) / 256, 256, 0, stream>>>(delta, (long)ME_ * DI_);

    scan_kernel<<<(B_ * DI_ * 16 + 127) / 128, 128, 0, stream>>>(
        delta, u, xdbl, xr, m_Alog + (long)i * DI_ * DS_, m_D + (long)i * DI_, y16);

    cvt(m_out_w + (long)i * DM_ * DI_, DI_, DI_, DM_, wbuf, DI_, DM_);
    launch_gemm(stream, y16, wbuf, nullptr, mo, nullptr, ME_, DM_, DI_, DI_, DI_, DM_, 1.f);
    add_residual_kernel<<<(ME_ * DM_ + 255) / 256, 256, 0, stream>>>(hf32, mo, hf16, (long)ME_ * DM_);

    // ---- cross-attention block ----
    const float* aw = a_in_w + (long)i * 3 * DM_ * DM_;
    const float* ab = a_in_b + (long)i * 3 * DM_;
    cvt(aw, DM_, DM_, DM_, wbuf, DM_, DM_);                       // Wq
    launch_gemm(stream, hf16, wbuf, ab, tmp, q16, ME_, DM_, DM_, DM_, DM_, DM_, 1.f);
    cvt(aw + (long)DM_ * DM_, DM_, DM_, DM_, wbuf, DM_, DM_);     // Wk
    launch_gemm(stream, enc16, wbuf, ab + DM_, tmp, k16, MENC_, DM_, DM_, DM_, DM_, DM_, 1.f);
    cvt(aw + 2L * DM_ * DM_, DM_, DM_, DM_, wbuf, DM_, DM_);      // Wv
    launch_gemm(stream, enc16, wbuf, ab + 2 * DM_, tmp, nullptr, MENC_, DM_, DM_, DM_, DM_, DM_, 1.f);
    transpose_v_kernel<<<(B_ * H_ * HD_ * LE_ + 255) / 256, 256, 0, stream>>>(tmp, vt16);

    // scores[b,h] = (Q_bh @ K_bh^T) * scale   (batched over z = b*H + h)
    launch_gemm(stream, q16, k16, nullptr, S, nullptr,
                L_, LE_, HD_, DM_, DM_, LE_, attn_scale,
                B_ * H_, H_,
                (long)L_ * DM_, (long)HD_,            // A: per-b, per-h
                (long)LE_ * DM_, (long)HD_,           // B: per-b, per-h
                (long)H_ * L_ * LE_, (long)L_ * LE_); // C: per-b, per-h
    softmax_kernel<<<B_ * H_ * L_, 256, 0, stream>>>(S, P16, LE_);

    // O[b,:,h,:] = P_bh @ V_bh  (V pre-transposed -> same A x B^T kernel)
    launch_gemm(stream, P16, vt16, nullptr, tmp, o16,
                L_, HD_, LE_, LE_, LE_, DM_, 1.f,
                B_ * H_, H_,
                (long)H_ * L_ * LE_, (long)L_ * LE_,
                (long)H_ * HD_ * LE_, (long)HD_ * LE_,
                (long)L_ * DM_, (long)HD_);

    cvt(a_out_w + (long)i * DM_ * DM_, DM_, DM_, DM_, wbuf, DM_, DM_);
    launch_gemm(stream, o16, wbuf, a_out_b + (long)i * DM_, mo, nullptr,
                ME_, DM_, DM_, DM_, DM_, DM_, 1.f);
    add_layernorm_kernel<<<ME_, 256, 0, stream>>>(hf32, mo, a_ln_w + (long)i * DM_,
                                                  a_ln_b + (long)i * DM_, hf16);
  }

  // ---- final norm + output projection (N padded 80 -> 128, then narrow) ----
  rmsnorm_kernel<<<ME_, 256, 0, stream>>>(hf32, normf_w, hn16);
  cvt(W_out, DM_, DM_, NM_, wbuf, DM_, NOP_);   // rows 80..127 zero-padded
  launch_gemm(stream, hn16, wbuf, nullptr, tmp, nullptr, ME_, NOP_, DM_, DM_, DM_, NOP_, 1.f);
  copy_out_kernel<<<(ME_ * NM_ + 255) / 256, 256, 0, stream>>>(tmp, out, (long)ME_ * NM_);
}
